// DTS_SNN_1D_18648747999262
// MI455X (gfx1250) — matrix operand order repaired and checked by reading, not compiled, not run on hardware
//
#include <hip/hip_runtime.h>
#include <hip/hip_bf16.h>
#include <stdint.h>

// ---------------------------------------------------------------------------
// DTS-SNN forward, MI455X (gfx1250, wave32, WMMA).
// 16 blocks x 512 threads; block b owns batch rows [16b, 16b+16) for all T.
// Hidden GEMM: v_wmma_f32_16x16x32_bf16, spikes are exact in bf16.
// ---------------------------------------------------------------------------

typedef __attribute__((ext_vector_type(16))) __bf16 v16bf;
typedef __attribute__((ext_vector_type(8)))  float  v8f;

#define T_STEPS 100
#define C_IN    768
#define PADC    784            // 768 + 2*8 padding for the radius-8 surface
#define GRP     192            // 768 / 4
#define NIN     3264           // 17 * 192
#define NHID    1024
#define NOUT    20
#define ROWS    16             // batch rows per block (WMMA M)
#define NTHR    512            // 16 waves
#define THRESH  0.3f
#define D1 0.951229424500714f      // exp(-1/20)  (tau_tr1, also tau_m)
#define D2 0.9834714538216174f     // exp(-1/60)  (tau_tr2)
#define DM 0.951229424500714f      // exp(-1/20)
#define DSY 0.8187307530779818f    // exp(-1/5)   (tau_s)

// LDS layout (bytes); all sections 16B aligned.
#define S_SURF_OFF 0
#define S_SIN_OFF  (ROWS*PADC*4)                   // 50176
#define S_SH_OFF   (S_SIN_OFF + ROWS*NIN*2)        // 154624
#define S_WOUT_OFF (S_SH_OFF + ROWS*NHID*4)        // 220160
#define S_DOT_OFF  (S_WOUT_OFF + NOUT*NHID*4)      // 302080
#define SMEM_BYTES (S_DOT_OFF + 320*4)             // 303360  (< 320KB WGP LDS)

union Frag16 { v16bf v; uint4 u[2]; };

// --- pre-pass: round-to-nearest-even f32 -> bf16 for w_hid ------------------
__global__ void to_bf16_kernel(const float* __restrict__ in,
                               unsigned short* __restrict__ out, int n) {
    int i = blockIdx.x * 256 + threadIdx.x;
    if (i < n) {
        unsigned u = __float_as_uint(in[i]);
        out[i] = (unsigned short)((u + 0x7FFFu + ((u >> 16) & 1u)) >> 16);
    }
}

__global__ __launch_bounds__(NTHR, 1)
void snn_forward(const float* __restrict__ events,
                 const float* __restrict__ w_enc,
                 const float* __restrict__ w_out,
                 const unsigned short* __restrict__ whid,   // bf16 bits, [1024][3264]
                 float* __restrict__ out)
{
    extern __shared__ char smem[];
    float*          s_surf = (float*)(smem + S_SURF_OFF);          // [16][784]
    unsigned short* s_sin  = (unsigned short*)(smem + S_SIN_OFF);  // [16][3264] bf16
    float*          s_sh   = (float*)(smem + S_SH_OFF);           // [16][1024]
    float*          s_wout = (float*)(smem + S_WOUT_OFF);         // [20][1024]
    float*          s_dot  = (float*)(smem + S_DOT_OFF);          // [320]

    const int tid   = threadIdx.x;
    const int bBase = blockIdx.x * ROWS;

    // zero surface (incl. pad), cache w_out in LDS
    for (int j = tid; j < ROWS * PADC; j += NTHR) s_surf[j] = 0.f;
    for (int j = tid; j < NOUT * NHID; j += NTHR) s_wout[j] = w_out[j];

    const float we0 = w_enc[0], we1 = w_enc[1], we2 = w_enc[2], we3 = w_enc[3];

    // per-thread recurrent state (registers)
    float tr1[24], tr2[24];        // 16*768 / 512 traces
    float m_in[102];               // 16*3264 / 512 input-LIF membranes
#pragma unroll
    for (int k = 0; k < 24; ++k) { tr1[k] = 0.f; tr2[k] = 0.f; }
#pragma unroll
    for (int k = 0; k < 102; ++k) m_in[k] = 0.f;

    v8f ch[4] = {};                // hidden synaptic current, WMMA C/D layout
    v8f mh[4] = {};                // hidden membrane, same layout
    float co = 0.f, mo = 0.f, ssum = 0.f;   // output state (tid < 320)

    const int lane = tid & 31;
    const int wv   = tid >> 5;     // wave id: owns hidden cols [64*wv, 64*wv+64)
    const int lh   = lane >> 4;    // lane half (WMMA fragment layout)
    const int lm   = lane & 15;
    const int nb   = wv * 64;

    __syncthreads();

    for (int t = 0; t < T_STEPS; ++t) {
        // ---- phase 1: dual traces + temporal surface -> LDS ----------------
#pragma unroll
        for (int k = 0; k < 24; ++k) {
            int e = tid + k * NTHR;            // < 12288 = 16*768
            int b = e / C_IN;
            int c = e - b * C_IN;
            size_t gi = ((size_t)(bBase + b) * T_STEPS + t) * C_IN + c;
            float ev = events[gi];
            if (t + 1 < T_STEPS) __builtin_prefetch(&events[gi + C_IN], 0, 0);
            ev = fminf(fmaxf(ev, 0.f), 1.f);
            tr1[k] = tr1[k] * D1 + ev;
            tr2[k] = tr2[k] * D2 + ev;
            s_surf[b * PADC + 8 + c] = (tr1[k] - tr2[k]) * 0.5f;
        }
        __syncthreads();

        // ---- phase 2: radius-8 encoder + input LIF -> spikes (bf16 bits) ---
#pragma unroll
        for (int k = 0; k < 102; ++k) {
            int e = tid + k * NTHR;            // e = b*3264 + i
            int b = e / NIN;
            int i = e - b * NIN;
            int r = i / GRP;                   // i = r*192 + g
            int g = i - r * GRP;
            const float* sp = &s_surf[b * PADC + (g << 2) + r];
            float enc = sp[0]*we0 + sp[1]*we1 + sp[2]*we2 + sp[3]*we3;
            float mp  = m_in[k];
            float spk = (mp > THRESH) ? 1.f : 0.f;   // previous spike, recomputed
            float m   = mp * DM * (1.f - spk) + enc;
            m_in[k] = m;
            s_sin[e] = (m > THRESH) ? (unsigned short)0x3F80 : (unsigned short)0;
        }
        __syncthreads();

        // ---- phase 3: hidden GEMM (16x1024, K=3264) via WMMA + hidden LIF --
        if (tid < 320) s_dot[tid] = 0.f;       // clear output-dot scratch
        {
            // fold synaptic decay into the first accumulator of the chain
            v8f acc0 = ch[0] * DSY, acc1 = ch[1] * DSY,
                acc2 = ch[2] * DSY, acc3 = ch[3] * DSY;
            const char* sinb = (const char*)s_sin + lm * (NIN * 2);
#pragma unroll 2
            for (int kc = 0; kc < NIN / 32; ++kc) {
                // A fragment: row lm, 16 contiguous K from lh*16 (matches B)
                Frag16 a;
                const uint4* ap = (const uint4*)(sinb + (kc * 32 + lh * 16) * 2);
                a.u[0] = ap[0];
                a.u[1] = ap[1];
                // B fragments: col = ntile*16+lm, 16 contiguous K from w_hid row
                const unsigned short* wk = whid + (size_t)kc * 32 + lh * 16;
                Frag16 b0, b1, b2, b3;
                const uint4* bp0 = (const uint4*)(wk + (size_t)(nb +  0 + lm) * NIN);
                const uint4* bp1 = (const uint4*)(wk + (size_t)(nb + 16 + lm) * NIN);
                const uint4* bp2 = (const uint4*)(wk + (size_t)(nb + 32 + lm) * NIN);
                const uint4* bp3 = (const uint4*)(wk + (size_t)(nb + 48 + lm) * NIN);
                b0.u[0] = bp0[0]; b0.u[1] = bp0[1];
                b1.u[0] = bp1[0]; b1.u[1] = bp1[1];
                b2.u[0] = bp2[0]; b2.u[1] = bp2[1];
                b3.u[0] = bp3[0]; b3.u[1] = bp3[1];
                acc0 = __builtin_amdgcn_wmma_f32_16x16x32_bf16(false, a.v, false, b0.v, (short)0, acc0, false, false);
                acc1 = __builtin_amdgcn_wmma_f32_16x16x32_bf16(false, a.v, false, b1.v, (short)0, acc1, false, false);
                acc2 = __builtin_amdgcn_wmma_f32_16x16x32_bf16(false, a.v, false, b2.v, (short)0, acc2, false, false);
                acc3 = __builtin_amdgcn_wmma_f32_16x16x32_bf16(false, a.v, false, b3.v, (short)0, acc3, false, false);
            }
            ch[0] = acc0; ch[1] = acc1; ch[2] = acc2; ch[3] = acc3;
        }
        // hidden LIF on C/D fragment layout: elem r -> row r+8*lh, col lm
#pragma unroll
        for (int tt = 0; tt < 4; ++tt) {
#pragma unroll
            for (int r = 0; r < 8; ++r) {
                float mp  = mh[tt][r];
                float spk = (mp > THRESH) ? 1.f : 0.f;
                float m   = mp * DM * (1.f - spk) + ch[tt][r];
                mh[tt][r] = m;
                s_sh[(r + lh * 8) * NHID + nb + tt * 16 + lm] =
                    (m > THRESH) ? 1.f : 0.f;
            }
        }
        __syncthreads();

        // ---- phase 4a: output layer partial dots (K=1024 split in 8 segs) --
#pragma unroll
        for (int q = 0; q < 5; ++q) {
            int task = tid + q * NTHR;          // < 2560 = 16*20*8
            int b    = task / 160;
            int rem  = task - b * 160;
            int o    = rem >> 3;
            int seg  = rem & 7;
            const float* shp = &s_sh[b * NHID + seg * 128];
            const float* wop = &s_wout[o * NHID + seg * 128];
            float p = 0.f;
#pragma unroll 16
            for (int j = 0; j < 128; ++j) p = fmaf(shp[j], wop[j], p);
            atomicAdd(&s_dot[b * NOUT + o], p);  // ds_add_f32
        }
        __syncthreads();

        // ---- phase 4b: output LIF + spike accumulation ---------------------
        if (tid < 320) {
            co = co * DSY + s_dot[tid];
            float spk = (mo > THRESH) ? 1.f : 0.f;
            mo = mo * DM * (1.f - spk) + co;
            ssum += (mo > THRESH) ? 1.f : 0.f;
        }
    }

    if (tid < 320) {
        int b = tid / NOUT;
        int o = tid - b * NOUT;
        out[(bBase + b) * NOUT + o] = ssum * (1.0f / T_STEPS);
    }
}

extern "C" void kernel_launch(void* const* d_in, const int* in_sizes, int n_in,
                              void* d_out, int out_size, void* d_ws, size_t ws_size,
                              hipStream_t stream) {
    const float* events = (const float*)d_in[0];   // [256][100][768] f32
    const float* w_enc  = (const float*)d_in[1];   // [4] f32
    const float* w_hid  = (const float*)d_in[2];   // [1024][3264] f32
    const float* w_out  = (const float*)d_in[3];   // [20][1024] f32
    // d_in[4] = batch_size scalar (256), unused

    unsigned short* whid_bf = (unsigned short*)d_ws;   // 6.7 MB bf16 weights
    const int nW = NHID * NIN;
    to_bf16_kernel<<<(nW + 255) / 256, 256, 0, stream>>>(w_hid, whid_bf, nW);

    hipFuncSetAttribute(reinterpret_cast<const void*>(snn_forward),
                        hipFuncAttributeMaxDynamicSharedMemorySize, SMEM_BYTES);
    snn_forward<<<256 / ROWS, NTHR, SMEM_BYTES, stream>>>(
        events, w_enc, w_out, whid_bf, (float*)d_out);
}
